// CausalRAGWithAdjustment_78520592105861
// MI455X (gfx1250) — matrix operand order repaired and checked by reading, not compiled, not run on hardware
//
#include <hip/hip_runtime.h>

#define B_ROWS   1024
#define N_CORPUS 200000
#define E_DIM    512
#define H_DIM    512
#define TOPK     8
#define CHUNK    1024
#define NCHUNKS  ((N_CORPUS + CHUNK - 1) / CHUNK)   // 196
#define RENC_K   (E_DIM * TOPK)                      // 4096
#define COMB     1536

typedef __attribute__((ext_vector_type(16))) __bf16 v16bf_t;
typedef __attribute__((ext_vector_type(8)))  float  v8f_t;

union BV { unsigned u[8]; v16bf_t v; };

__device__ __forceinline__ unsigned short f2bf(float f) {
  unsigned u = __float_as_uint(f);
  u += 0x7fffu + ((u >> 16) & 1u);          // round-to-nearest-even
  return (unsigned short)(u >> 16);
}

// Branchless insert of (v,id) into a descending top-8 list, (score desc, idx asc).
// If v does not qualify, the swap chain performs no swaps -> always correct.
// Lowers to v_cmp + v_cndmask only (no EXEC manipulation).
__device__ __forceinline__ void tk_insert(float v, int id, float* s, int* ix) {
  float cv = v; int ci = id;
#pragma unroll
  for (int j = 0; j < TOPK; ++j) {
    const bool better = (cv > s[j]) || (cv == s[j] && ci < ix[j]);
    const float ns = better ? cv : s[j];
    const int   ni = better ? ci : ix[j];
    cv = better ? s[j] : cv;
    ci = better ? ix[j] : ci;
    s[j] = ns; ix[j] = ni;
  }
}

// Generic 16x16 tile GEMM over K (multiple of 32), bf16 inputs, f32 accum.
// A: row-major [16 x K] (row stride lda). Bt: row-major [16 x K], rows = output columns.
__device__ __forceinline__ void wmma_gemm_tile(
    const unsigned short* __restrict__ A, int lda,
    const unsigned short* __restrict__ Bt, int ldb,
    int K, v8f_t& acc)
{
  const int lane = threadIdx.x & 31;
  const int m  = lane & 15;
  const int hi = lane >> 4;
  for (int kb = 0; kb < K; kb += 32) {
    BV a, b;
#pragma unroll
    for (int d = 0; d < 8; ++d) {
      const int ka = kb + ((d & 3) << 1) + ((d >> 2) << 4) + (hi << 3);
      a.u[d] = *(const unsigned*)(A + m * lda + ka);
      const int kk = kb + ((d & 3) << 1) + ((d >> 2) << 4) + (hi << 3);
      b.u[d] = *(const unsigned*)(Bt + m * ldb + kk);
    }
    acc = __builtin_amdgcn_wmma_f32_16x16x32_bf16(false, a.v, false, b.v,
                                                  (short)0, acc, false, false);
  }
}

// ---------------- corpus: l2-normalize rows, write bf16 ----------------
__global__ void k_norm_corpus(const float* __restrict__ corpus,
                              unsigned short* __restrict__ cbf) {
  const int wave = threadIdx.x >> 5, lane = threadIdx.x & 31;
  const int row = blockIdx.x * 8 + wave;
  if (row >= N_CORPUS) return;
  const float* src = corpus + (size_t)row * E_DIM;
  float vals[16]; float ss = 0.f;
#pragma unroll
  for (int i = 0; i < 16; ++i) { const float v = src[lane + (i << 5)]; vals[i] = v; ss += v * v; }
#pragma unroll
  for (int off = 16; off; off >>= 1) ss += __shfl_xor(ss, off, 32);
  const float inv = 1.f / fmaxf(sqrtf(ss), 1e-12f);
  unsigned short* dst = cbf + (size_t)row * E_DIM;
#pragma unroll
  for (int i = 0; i < 16; ++i) dst[lane + (i << 5)] = f2bf(vals[i] * inv);
}

// ---------------- q = l2norm(patient @ W_pe + b_pe), bf16 ----------------
__global__ void k_encode_patient(const float* __restrict__ patient,
                                 const float* __restrict__ Wpe,
                                 const float* __restrict__ bpe,
                                 unsigned short* __restrict__ qbf) {
  __shared__ float prow[68];
  __shared__ float red[256];
  const int b = blockIdx.x, t = threadIdx.x;
  if (t < 66) prow[t] = patient[b * 66 + t];
  __syncthreads();
  float acc[2];
#pragma unroll
  for (int j = 0; j < 2; ++j) {
    const int e = t + (j << 8);
    float a = bpe[e];
    for (int k = 0; k < 66; ++k) a += prow[k] * Wpe[k * E_DIM + e];
    acc[j] = a;
  }
  red[t] = acc[0] * acc[0] + acc[1] * acc[1];
  __syncthreads();
  for (int s = 128; s; s >>= 1) { if (t < s) red[t] += red[t + s]; __syncthreads(); }
  const float inv = 1.f / fmaxf(sqrtf(red[0]), 1e-12f);
#pragma unroll
  for (int j = 0; j < 2; ++j) qbf[(size_t)b * E_DIM + t + (j << 8)] = f2bf(acc[j] * inv);
}

// ---------------- transpose+convert weights: out[n*K+k] = bf16(in[k*N+n]) ----------------
__global__ void k_transpose_bf16(const float* __restrict__ in,
                                 unsigned short* __restrict__ out, int K, int N) {
  const int i = blockIdx.x * blockDim.x + threadIdx.x;
  if (i >= K * N) return;
  const int n = i / K, k = i - n * K;
  out[i] = f2bf(in[k * N + n]);
}

// ---------------- fused sim GEMM + per-chunk top-8 ----------------
// grid: (B/64, NCHUNKS), block 256 (8 waves). Each wave: 64x16 output slab (4 M-tiles,
// one shared B tile -> 4x corpus reuse), running per-row top-8 in registers with a
// threshold-mask filter so the insertion chain runs only for qualifying candidates.
__global__ void __launch_bounds__(256) k_sim_topk(
    const unsigned short* __restrict__ qbf, const unsigned short* __restrict__ cbf,
    float* __restrict__ pscore, int* __restrict__ pidx)
{
  __shared__ float slab[8][16][17];
  __shared__ float shs[64][64];
  __shared__ int   shi[64][64];
  const int wave = threadIdx.x >> 5, lane = threadIdx.x & 31;
  const int m = lane & 15, hi = lane >> 4, col = lane & 15;
  const int rowBase = blockIdx.x << 6;
  const int chunk0 = blockIdx.y * CHUNK;
  const int chunkEnd = (chunk0 + CHUNK < N_CORPUS) ? (chunk0 + CHUNK) : N_CORPUS;

  float ts[4][TOPK]; int ti[4][TOPK];
#pragma unroll
  for (int t = 0; t < 4; ++t)
#pragma unroll
    for (int j = 0; j < TOPK; ++j) { ts[t][j] = -3.0e38f; ti[t][j] = 0x7fffffff; }

  const unsigned short* A = qbf + (size_t)rowBase * E_DIM;

  for (int n0 = chunk0 + (wave << 4); n0 < chunkEnd; n0 += 128) {
    v8f_t acc[4];
#pragma unroll
    for (int t = 0; t < 4; ++t) acc[t] = (v8f_t){0.f,0.f,0.f,0.f,0.f,0.f,0.f,0.f};
    const unsigned short* Bt = cbf + (size_t)n0 * E_DIM;
#pragma unroll 1
    for (int kb = 0; kb < E_DIM; kb += 32) {
      BV bm;
#pragma unroll
      for (int d = 0; d < 8; ++d)
        bm.u[d] = *(const unsigned*)(Bt + m * E_DIM + (kb + ((d & 3) << 1) + ((d >> 2) << 4) + (hi << 3)));
      BV am[4];
#pragma unroll
      for (int d = 0; d < 8; ++d) {
        const int ka = kb + ((d & 3) << 1) + ((d >> 2) << 4) + (hi << 3);
#pragma unroll
        for (int t = 0; t < 4; ++t)
          am[t].u[d] = *(const unsigned*)(A + ((t << 4) + m) * E_DIM + ka);
      }
#pragma unroll
      for (int t = 0; t < 4; ++t)
        acc[t] = __builtin_amdgcn_wmma_f32_16x16x32_bf16(false, am[t].v, false, bm.v,
                                                         (short)0, acc[t], false, false);
    }
    // per-tile selection (wave-private LDS slab; single instruction stream -> no barrier)
#pragma unroll
    for (int t = 0; t < 4; ++t) {
#pragma unroll
      for (int j = 0; j < 8; ++j) slab[wave][j + (hi << 3)][col] = acc[t][j];
      if (lane < 16) {
        const float thr = ts[t][TOPK - 1];
        unsigned msk = 0u;
#pragma unroll
        for (int c = 0; c < 16; ++c)
          msk |= (slab[wave][lane][c] > thr) ? (1u << c) : 0u;
        while (msk) {
          const int c = __ffs(msk) - 1;
          msk &= msk - 1u;
          tk_insert(slab[wave][lane][c], n0 + c, ts[t], ti[t]);
        }
      }
    }
  }

  if (lane < 16) {
#pragma unroll
    for (int t = 0; t < 4; ++t)
#pragma unroll
      for (int j = 0; j < TOPK; ++j) {
        shs[(t << 4) + lane][wave * TOPK + j] = ts[t][j];
        shi[(t << 4) + lane][wave * TOPK + j] = ti[t][j];
      }
  }
  __syncthreads();
  // merge the 8 waves' disjoint-column top-8s: one thread per row (64 rows)
  if (threadIdx.x < 64) {
    const int r = threadIdx.x;
    float fs[TOPK]; int fi[TOPK];
#pragma unroll
    for (int j = 0; j < TOPK; ++j) { fs[j] = -3.0e38f; fi[j] = 0x7fffffff; }
    for (int c = 0; c < 64; ++c) {
      const float v = shs[r][c];
      if (v >= fs[TOPK - 1]) tk_insert(v, shi[r][c], fs, fi);
    }
    const size_t base = ((size_t)(rowBase + r) * NCHUNKS + blockIdx.y) * TOPK;
#pragma unroll
    for (int j = 0; j < TOPK; ++j) { pscore[base + j] = fs[j]; pidx[base + j] = fi[j]; }
  }
}

// ---------------- final top-8 across chunks ----------------
__global__ void k_topk_final(const float* __restrict__ pscore, const int* __restrict__ pidx,
                             float* __restrict__ out_scores, float* __restrict__ out_idx_f,
                             int* __restrict__ topidx)
{
  __shared__ float ms[32][TOPK];
  __shared__ int   mi[32][TOPK];
  const int b = blockIdx.x, lane = threadIdx.x;
  const int total = NCHUNKS * TOPK;   // 1568
  float s[TOPK]; int ix[TOPK];
#pragma unroll
  for (int j = 0; j < TOPK; ++j) { s[j] = -3.0e38f; ix[j] = 0x7fffffff; }
  const float* ps = pscore + (size_t)b * total;
  const int*   pi = pidx + (size_t)b * total;
  for (int c = lane; c < total; c += 32) {
    const float v = ps[c];
    if (v >= s[TOPK - 1]) tk_insert(v, pi[c], s, ix);
  }
#pragma unroll
  for (int j = 0; j < TOPK; ++j) { ms[lane][j] = s[j]; mi[lane][j] = ix[j]; }
  __syncthreads();
  if (lane == 0) {
    float fs[TOPK]; int fi[TOPK];
#pragma unroll
    for (int j = 0; j < TOPK; ++j) { fs[j] = -3.0e38f; fi[j] = 0x7fffffff; }
    for (int l = 0; l < 32; ++l)
#pragma unroll
      for (int j = 0; j < TOPK; ++j) {
        const float v = ms[l][j];
        if (v >= fs[TOPK - 1]) tk_insert(v, mi[l][j], fs, fi);
      }
#pragma unroll
    for (int j = 0; j < TOPK; ++j) {
      out_scores[b * TOPK + j] = fs[j];
      out_idx_f[b * TOPK + j]  = (float)fi[j];
      topidx[b * TOPK + j]     = fi[j];
    }
  }
}

// ---------------- gather retrieved raw corpus rows -> bf16 flat [B, K*E] ----------------
__global__ void k_gather(const float* __restrict__ corpus, const int* __restrict__ topidx,
                         unsigned short* __restrict__ flatbf)
{
  const int wave = threadIdx.x >> 5, lane = threadIdx.x & 31;
  const int b = blockIdx.x;
  const int r = topidx[b * TOPK + wave];
  const float* src = corpus + (size_t)r * E_DIM;
  unsigned short* dst = flatbf + (size_t)b * RENC_K + wave * E_DIM;
#pragma unroll
  for (int i = 0; i < 16; ++i) dst[lane + (i << 5)] = f2bf(src[lane + (i << 5)]);
}

// ---------------- t_enc, c_enc -> combined[:, 0:1024) ----------------
__global__ void k_enc_tc(const float* __restrict__ treat, const float* __restrict__ conf,
                         const float* __restrict__ Wte, const float* __restrict__ bte,
                         const float* __restrict__ Wce, const float* __restrict__ bce,
                         unsigned short* __restrict__ combined)
{
  __shared__ float trow[2];
  __shared__ float crow[64];
  const int b = blockIdx.x, t = threadIdx.x;
  if (t < 2)  trow[t] = treat[b * 2 + t];
  if (t < 64) crow[t] = conf[b * 64 + t];
  __syncthreads();
  unsigned short* outr = combined + (size_t)b * COMB;
#pragma unroll
  for (int j = 0; j < 2; ++j) {
    const int e = t + (j << 8);
    const float te = bte[e] + trow[0] * Wte[e] + trow[1] * Wte[E_DIM + e];
    float ce = bce[e];
    for (int k = 0; k < 64; ++k) ce += crow[k] * Wce[k * E_DIM + e];
    outr[e] = f2bf(te);
    outr[E_DIM + e] = f2bf(ce);
  }
}

// ---------------- generic WMMA GEMM: one wave per 16x16 tile ----------------
__global__ void k_wmma_gemm(const unsigned short* __restrict__ A, int lda,
                            const unsigned short* __restrict__ Bt, int ldb, int K,
                            const float* __restrict__ bias,
                            unsigned short* __restrict__ out_bf, float* __restrict__ out_f32,
                            int ldo, int relu)
{
  const int lane = threadIdx.x & 31;
  const int rowTile = blockIdx.x << 4;
  const int n0 = blockIdx.y << 4;
  v8f_t acc = (v8f_t){0.f,0.f,0.f,0.f,0.f,0.f,0.f,0.f};
  wmma_gemm_tile(A + (size_t)rowTile * lda, lda, Bt + (size_t)n0 * ldb, ldb, K, acc);
  const int hi = lane >> 4, col = lane & 15;
  const int n = n0 + col;
  const float bv = bias[n];
#pragma unroll
  for (int j = 0; j < 8; ++j) {
    const int row = rowTile + j + (hi << 3);
    float v = acc[j] + bv;
    if (relu) v = fmaxf(v, 0.f);
    if (out_bf) out_bf[(size_t)row * ldo + n] = f2bf(v);
    else        out_f32[(size_t)row * ldo + n] = v;
  }
}

// ---------------- outcome = h2 @ W_o3 + b_o3 ----------------
__global__ void k_outcome(const float* __restrict__ h2, const float* __restrict__ Wo3,
                          const float* __restrict__ bo3, float* __restrict__ outc)
{
  const int wave = threadIdx.x >> 5, lane = threadIdx.x & 31;
  const int b = blockIdx.x * 8 + wave;
  const float* hr = h2 + (size_t)b * 256;
  float s = 0.f;
#pragma unroll
  for (int i = 0; i < 8; ++i) s += hr[lane + (i << 5)] * Wo3[lane + (i << 5)];
#pragma unroll
  for (int off = 16; off; off >>= 1) s += __shfl_xor(s, off, 32);
  if (lane == 0) outc[b] = s + bo3[0];
}

// ---------------- propensity head (fp32 VALU) ----------------
__global__ void k_prop(const float* __restrict__ conf, const float* __restrict__ Wp1,
                       const float* __restrict__ bp1, const float* __restrict__ Wp2,
                       const float* __restrict__ bp2, float* __restrict__ prop)
{
  __shared__ float crow[64];
  __shared__ float r0[256], r1[256];
  const int b = blockIdx.x, t = threadIdx.x;
  if (t < 64) crow[t] = conf[b * 64 + t];
  __syncthreads();
  float l0 = 0.f, l1 = 0.f;
#pragma unroll
  for (int j = 0; j < 2; ++j) {
    const int e = t + (j << 8);
    float p = bp1[e];
    for (int k = 0; k < 64; ++k) p += crow[k] * Wp1[k * E_DIM + e];
    p = fmaxf(p, 0.f);
    l0 += p * Wp2[e * 2 + 0];
    l1 += p * Wp2[e * 2 + 1];
  }
  r0[t] = l0; r1[t] = l1;
  __syncthreads();
  for (int s = 128; s; s >>= 1) { if (t < s) { r0[t] += r0[t + s]; r1[t] += r1[t + s]; } __syncthreads(); }
  if (t == 0) {
    const float a = r0[0] + bp2[0], c = r1[0] + bp2[1];
    const float mx = fmaxf(a, c);
    const float ea = __expf(a - mx), ec = __expf(c - mx);
    const float inv = 1.f / (ea + ec);
    prop[b * 2 + 0] = ea * inv;
    prop[b * 2 + 1] = ec * inv;
  }
}

extern "C" void kernel_launch(void* const* d_in, const int* in_sizes, int n_in,
                              void* d_out, int out_size, void* d_ws, size_t ws_size,
                              hipStream_t stream) {
  (void)in_sizes; (void)n_in; (void)out_size; (void)ws_size;
  const float* treatment = (const float*)d_in[0];
  const float* conf      = (const float*)d_in[1];
  const float* patient   = (const float*)d_in[2];
  const float* corpus    = (const float*)d_in[3];
  const float* Wpe = (const float*)d_in[4];  const float* bpe = (const float*)d_in[5];
  const float* Wte = (const float*)d_in[6];  const float* bte = (const float*)d_in[7];
  const float* Wce = (const float*)d_in[8];  const float* bce = (const float*)d_in[9];
  const float* Wre = (const float*)d_in[10]; const float* bre = (const float*)d_in[11];
  const float* Wo1 = (const float*)d_in[12]; const float* bo1 = (const float*)d_in[13];
  const float* Wo2 = (const float*)d_in[14]; const float* bo2 = (const float*)d_in[15];
  const float* Wo3 = (const float*)d_in[16]; const float* bo3 = (const float*)d_in[17];
  const float* Wp1 = (const float*)d_in[18]; const float* bp1 = (const float*)d_in[19];
  const float* Wp2 = (const float*)d_in[20]; const float* bp2 = (const float*)d_in[21];

  float* out         = (float*)d_out;
  float* out_outcome = out;                      // [1024]
  float* out_scores  = out + B_ROWS;             // [1024*8]
  float* out_idx     = out_scores + B_ROWS * TOPK;
  float* out_prop    = out_idx + B_ROWS * TOPK;  // [1024*2]

  char* p = (char*)d_ws;
  auto carve = [&](size_t bytes) { char* r = p; p += (bytes + 255) & ~(size_t)255; return r; };
  unsigned short* cbf   = (unsigned short*)carve((size_t)N_CORPUS * E_DIM * 2);
  unsigned short* qbf   = (unsigned short*)carve((size_t)B_ROWS * E_DIM * 2);
  float*          psc   = (float*)carve((size_t)B_ROWS * NCHUNKS * TOPK * 4);
  int*            pix   = (int*)carve((size_t)B_ROWS * NCHUNKS * TOPK * 4);
  int*            tix   = (int*)carve((size_t)B_ROWS * TOPK * 4);
  unsigned short* flatb = (unsigned short*)carve((size_t)B_ROWS * RENC_K * 2);
  unsigned short* comb  = (unsigned short*)carve((size_t)B_ROWS * COMB * 2);
  unsigned short* h1b   = (unsigned short*)carve((size_t)B_ROWS * H_DIM * 2);
  float*          h2f   = (float*)carve((size_t)B_ROWS * 256 * 4);
  unsigned short* WtRe  = (unsigned short*)carve((size_t)H_DIM * RENC_K * 2);
  unsigned short* WtO1  = (unsigned short*)carve((size_t)H_DIM * COMB * 2);
  unsigned short* WtO2  = (unsigned short*)carve((size_t)256 * H_DIM * 2);

  k_norm_corpus<<<N_CORPUS / 8, 256, 0, stream>>>(corpus, cbf);
  k_encode_patient<<<B_ROWS, 256, 0, stream>>>(patient, Wpe, bpe, qbf);
  k_transpose_bf16<<<(RENC_K * H_DIM + 255) / 256, 256, 0, stream>>>(Wre, WtRe, RENC_K, H_DIM);
  k_transpose_bf16<<<(COMB * H_DIM + 255) / 256, 256, 0, stream>>>(Wo1, WtO1, COMB, H_DIM);
  k_transpose_bf16<<<(H_DIM * 256 + 255) / 256, 256, 0, stream>>>(Wo2, WtO2, H_DIM, 256);

  k_sim_topk<<<dim3(B_ROWS / 64, NCHUNKS), 256, 0, stream>>>(qbf, cbf, psc, pix);
  k_topk_final<<<B_ROWS, 32, 0, stream>>>(psc, pix, out_scores, out_idx, tix);
  k_gather<<<B_ROWS, 256, 0, stream>>>(corpus, tix, flatb);

  k_enc_tc<<<B_ROWS, 256, 0, stream>>>(treatment, conf, Wte, bte, Wce, bce, comb);
  // r_enc -> combined[:, 1024:1536)
  k_wmma_gemm<<<dim3(B_ROWS / 16, H_DIM / 16), 32, 0, stream>>>(
      flatb, RENC_K, WtRe, RENC_K, RENC_K, bre, comb + 1024, nullptr, COMB, 0);
  // h1 = relu(combined @ W_o1 + b_o1)
  k_wmma_gemm<<<dim3(B_ROWS / 16, H_DIM / 16), 32, 0, stream>>>(
      comb, COMB, WtO1, COMB, COMB, bo1, h1b, nullptr, H_DIM, 1);
  // h2 = relu(h1 @ W_o2 + b_o2), fp32 out
  k_wmma_gemm<<<dim3(B_ROWS / 16, 256 / 16), 32, 0, stream>>>(
      h1b, H_DIM, WtO2, H_DIM, H_DIM, bo2, nullptr, h2f, 256, 1);

  k_outcome<<<B_ROWS / 8, 256, 0, stream>>>(h2f, Wo3, bo3, out_outcome);
  k_prop<<<B_ROWS, 256, 0, stream>>>(conf, Wp1, bp1, Wp2, bp2, out_prop);
}
